// SequenceBlock_43404939493956
// MI455X (gfx1250) — compile-verified, matrix operand order checked by reading, not run on hardware
//
#include <hip/hip_runtime.h>
#include <hip/hip_bf16.h>
#include <math.h>

// ---------------------------------------------------------------------------
// SequenceBlock for MI455X (gfx1250, wave32, WMMA, async-to-LDS)
//   T=16384, HIDDEN=1024, HIPPO=512
//   Pipeline: LN(f32->f16) -> igates WMMA GEMM (LDS-staged A, 4-tile reg block)
//             -> 16-WG cooperative GRU scan (w_hh slices async-staged in LDS,
//                fp32 recurrence, atomic epoch barrier, fused GELU -> g f16)
//             -> fused dual WMMA GEMM (8 accumulators) + sigmoid gate + skip
// ---------------------------------------------------------------------------

#define T_LEN   16384
#define HID     1024
#define NH      512
#define G3      1536   // 3*NH
#define NWG     16     // cooperating workgroups in the scan

typedef __attribute__((ext_vector_type(16))) _Float16 v16h;
typedef __attribute__((ext_vector_type(8)))  _Float16 v8h;
typedef __attribute__((ext_vector_type(8)))  float    v8f;

// Padded LDS strides (halves): +8 halves = 16B keeps 16-B alignment for b128
// ds loads while rotating banks by 4/row -> 16 fragment lanes hit 16 banks.
#define A_STRIDE_IG  (HID + 8)   // 1032 halves
#define A_STRIDE_OUT (NH + 8)    // 520 halves

// Per ISA 7.12.2 (f16 16x16x32): each lane holds two contiguous 8-half chunks
// at k = off and k = 16+off, off = (lane&16)?8:0.  Works for A rows and, since
// our B matrices are (N,K) row-major, for B columns too.
__device__ __forceinline__ v16h load_frag16(const _Float16* base, int off) {
  v8h lo = *(const v8h*)(base + off);
  v8h hi = *(const v8h*)(base + 16 + off);
  v16h r;
#pragma unroll
  for (int i = 0; i < 8; ++i) { r[i] = lo[i]; r[8 + i] = hi[i]; }
  return r;
}

// CDNA5 async copy: global -> LDS, 16B per lane, tracked by ASYNCcnt.
// (generic LDS pointer low 32 bits == LDS byte address per flat aperture rule)
__device__ __forceinline__ void async_g2l_b128(unsigned lds_byte, const void* gptr) {
  asm volatile("global_load_async_to_lds_b128 %0, %1, off"
               :: "v"(lds_byte), "v"(gptr)
               : "memory");
}
__device__ __forceinline__ void wait_async0() {
  asm volatile("s_wait_asynccnt 0x0" ::: "memory");
}

__device__ __forceinline__ float sigmoidf_(float x) {
  return 1.0f / (1.0f + __expf(-x));
}

// ---------------------------------------------------------------------------
// Kernel 1: LayerNorm row (1024) -> f16, one block (256 thr) per row
// ---------------------------------------------------------------------------
__global__ void ln_to_f16(const float* __restrict__ x,
                          const float* __restrict__ w,
                          const float* __restrict__ b,
                          _Float16* __restrict__ xn) {
  const int t = blockIdx.x;
  const int tid = threadIdx.x;
  const float4 v = ((const float4*)(x + (size_t)t * HID))[tid];
  float s = v.x + v.y + v.z + v.w;
  float q = v.x * v.x + v.y * v.y + v.z * v.z + v.w * v.w;
#pragma unroll
  for (int sh = 16; sh; sh >>= 1) {
    s += __shfl_xor(s, sh);
    q += __shfl_xor(q, sh);
  }
  __shared__ float ss[8], qq[8];
  if ((tid & 31) == 0) { ss[tid >> 5] = s; qq[tid >> 5] = q; }
  __syncthreads();
  if (tid == 0) {
    float S = 0.f, Q = 0.f;
#pragma unroll
    for (int i = 0; i < 8; ++i) { S += ss[i]; Q += qq[i]; }
    ss[0] = S; qq[0] = Q;
  }
  __syncthreads();
  const float mu  = ss[0] * (1.0f / HID);
  const float var = qq[0] * (1.0f / HID) - mu * mu;
  const float inv = rsqrtf(var + 1e-5f);
  _Float16* orow = xn + (size_t)t * HID;
  const int c = tid * 4;
  orow[c]     = (_Float16)((v.x - mu) * inv * w[c]     + b[c]);
  orow[c + 1] = (_Float16)((v.y - mu) * inv * w[c + 1] + b[c + 1]);
  orow[c + 2] = (_Float16)((v.z - mu) * inv * w[c + 2] + b[c + 2]);
  orow[c + 3] = (_Float16)((v.w - mu) * inv * w[c + 3] + b[c + 3]);
}

// ---------------------------------------------------------------------------
// Kernel 2: f32 -> f16 weight conversion
// ---------------------------------------------------------------------------
__global__ void cvt_f16(const float* __restrict__ src, _Float16* __restrict__ dst, int n) {
  const int i = blockIdx.x * 256 + threadIdx.x;
  if (i < n) dst[i] = (_Float16)src[i];
}

// ---------------------------------------------------------------------------
// Kernel 3: igates = xn(16384x1024,f16) @ w_ih^T + b  -> f32 (T x 1536)
//   Block = 16-row M stripe (async-staged in LDS, shared by 8 waves);
//   each wave register-blocks 4 N tiles (16x64), reusing the A fragment.
//   grid (T/16, G3/512)
// ---------------------------------------------------------------------------
__global__ void igates_gemm(const _Float16* __restrict__ xn,
                            const _Float16* __restrict__ wih,
                            const float* __restrict__ bias,
                            float* __restrict__ ig) {
  __shared__ _Float16 Alds[16 * A_STRIDE_IG];
  const int tid  = threadIdx.x;
  const int wave = tid >> 5;
  const int lane = tid & 31;
  const int tM = blockIdx.x * 16;                       // [0, 16384)
  const int tN = blockIdx.y * 512 + wave * 64;          // [0, 1536)

  // async-stage the 16x1024 A stripe: 2048 x 16B chunks, 8 per thread
  {
    const unsigned abase = (unsigned)(uintptr_t)&Alds[0];
#pragma unroll
    for (int it = 0; it < 8; ++it) {
      const int idx = tid + it * 256;        // [0, 2048)
      const int row = idx >> 7;              // [0, 16)
      const int c   = idx & 127;             // 16B chunk within row
      async_g2l_b128(abase + (unsigned)(row * A_STRIDE_IG + c * 8) * 2u,
                     xn + (size_t)(tM + row) * HID + c * 8);
    }
  }
  wait_async0();
  __syncthreads();

  const int off = (lane & 16) ? 8 : 0;
  const _Float16* arow = &Alds[(lane & 15) * A_STRIDE_IG];
  const _Float16* brow0 = wih + (size_t)(tN +  0 + (lane & 15)) * HID;
  const _Float16* brow1 = wih + (size_t)(tN + 16 + (lane & 15)) * HID;
  const _Float16* brow2 = wih + (size_t)(tN + 32 + (lane & 15)) * HID;
  const _Float16* brow3 = wih + (size_t)(tN + 48 + (lane & 15)) * HID;

  v8f c0 = {}, c1 = {}, c2 = {}, c3 = {};
#pragma unroll 2
  for (int kb = 0; kb < HID; kb += 32) {
    const v16h a  = load_frag16(arow + kb, off);
    const v16h f0 = load_frag16(brow0 + kb, off);
    const v16h f1 = load_frag16(brow1 + kb, off);
    const v16h f2 = load_frag16(brow2 + kb, off);
    const v16h f3 = load_frag16(brow3 + kb, off);
    c0 = __builtin_amdgcn_wmma_f32_16x16x32_f16(false, a, false, f0, (short)0, c0, false, false);
    c1 = __builtin_amdgcn_wmma_f32_16x16x32_f16(false, a, false, f1, (short)0, c1, false, false);
    c2 = __builtin_amdgcn_wmma_f32_16x16x32_f16(false, a, false, f2, (short)0, c2, false, false);
    c3 = __builtin_amdgcn_wmma_f32_16x16x32_f16(false, a, false, f3, (short)0, c3, false, false);
  }

  const int N  = lane & 15;
  const int mb = (lane & 16) ? 8 : 0;
#pragma unroll
  for (int i = 0; i < 8; ++i) {
    const size_t rowoff = (size_t)(tM + i + mb) * G3;
    ig[rowoff + tN +  0 + N] = c0[i] + bias[tN +  0 + N];
    ig[rowoff + tN + 16 + N] = c1[i] + bias[tN + 16 + N];
    ig[rowoff + tN + 32 + N] = c2[i] + bias[tN + 32 + N];
    ig[rowoff + tN + 48 + N] = c3[i] + bias[tN + 48 + N];
  }
}

// ---------------------------------------------------------------------------
// Kernel 4: cooperative GRU scan, 16 co-resident workgroups.
//   WG j owns h[32j..32j+32); its 96 w_hh rows (r/z/n gates) are async-staged
//   into LDS once.  h broadcast via double-buffered global array + atomic
//   epoch counter.  Fuses GELU(tanh approx) and stores g as f16.
// ---------------------------------------------------------------------------
__global__ void gru_scan(const float* __restrict__ whh,
                         const float* __restrict__ igates,
                         const float* __restrict__ bn,
                         float* __restrict__ hbuf,          // 2 x 512
                         unsigned int* __restrict__ cnt,
                         _Float16* __restrict__ g_out,      // T x 512
                         float* __restrict__ hfinal) {      // 512
  extern __shared__ float lds[];
  float* wl = lds;                // 96*512 weights slice
  float* hc = lds + 96 * NH;      // 512 current h
  float* hg = hc + NH;            // 96 gate dots
  const int j  = blockIdx.x;      // 0..15
  const int i0 = j * 32;
  const int tid = threadIdx.x;
  const int lane = tid & 31, wave = tid >> 5;

  // Async-stage this WG's 96 rows of w_hh (rows gate*512 + i0 + e), 192 KB.
  {
    const unsigned wbase = (unsigned)(uintptr_t)wl;
    for (int idx = tid; idx < 96 * (NH / 4); idx += 256) {   // 16B chunks
      const int rr = idx >> 7;          // local row [0,96)
      const int c4 = idx & 127;
      const int grow = (rr / 32) * NH + i0 + (rr & 31);
      async_g2l_b128(wbase + (unsigned)(rr * NH + c4 * 4) * 4u,
                     whh + (size_t)grow * NH + c4 * 4);
    }
  }
  wait_async0();
  __syncthreads();

  for (int t = 0; t < T_LEN; ++t) {
    // pull full h (512 f32) from the shared double buffer (L2-resident);
    // normal loads: they must observe the agent-scope acquire below.
    for (int idx = tid; idx < NH / 4; idx += 256)
      ((float4*)hc)[idx] = ((const float4*)(hbuf + (size_t)(t & 1) * NH))[idx];
    __syncthreads();

    // 96 dot products: wave w handles rows [12w, 12w+12)
    for (int rr = wave * 12; rr < wave * 12 + 12; ++rr) {
      const float* wrow = wl + rr * NH;
      float acc = 0.f;
#pragma unroll
      for (int kk = 0; kk < 16; ++kk)
        acc = fmaf(wrow[lane + kk * 32], hc[lane + kk * 32], acc);
#pragma unroll
      for (int sh = 16; sh; sh >>= 1) acc += __shfl_xor(acc, sh);
      if (lane == 0) hg[rr] = acc;
    }
    __syncthreads();

    if (tid < 32) {
      const int i = i0 + tid;
      const float hr = hg[tid], hz = hg[32 + tid], hn = hg[64 + tid];
      const float* igt = igates + (size_t)t * G3;
      const float r = sigmoidf_(igt[i] + hr);
      const float z = sigmoidf_(igt[NH + i] + hz);
      const float n = tanhf(igt[2 * NH + i] + r * (hn + bn[i]));
      const float hold = hc[i];
      const float hnew = n + z * (hold - n);
      hbuf[(size_t)((t + 1) & 1) * NH + i] = hnew;
      // approximate (tanh) GELU, matching jax.nn.gelu default
      const float u = 0.7978845608028654f * (hnew + 0.044715f * hnew * hnew * hnew);
      g_out[(size_t)t * NH + i] = (_Float16)(0.5f * hnew * (1.0f + tanhf(u)));
      if (t == T_LEN - 1) hfinal[i] = hnew;
      __threadfence();
    }
    __syncthreads();
    if (tid == 0) {
      __hip_atomic_fetch_add(cnt, 1u, __ATOMIC_RELEASE, __HIP_MEMORY_SCOPE_AGENT);
      const unsigned target = (unsigned)NWG * (unsigned)(t + 1);
      while (__hip_atomic_load(cnt, __ATOMIC_ACQUIRE, __HIP_MEMORY_SCOPE_AGENT) < target)
        __builtin_amdgcn_s_sleep(1);
    }
    __syncthreads();
  }
}

// ---------------------------------------------------------------------------
// Kernel 5: out = skip + (g@out_w^T + out_b) * sigmoid(g@out2_w^T + out2_b)
//   16-row A stripe async-staged in LDS; each wave: 4 N tiles x 2 matrices
//   (8 WMMA accumulators).  grid (T/16, HID/512)
// ---------------------------------------------------------------------------
__global__ void out_gemm(const _Float16* __restrict__ g,
                         const _Float16* __restrict__ w1,
                         const _Float16* __restrict__ w2,
                         const float* __restrict__ b1,
                         const float* __restrict__ b2,
                         const float* __restrict__ skip,
                         float* __restrict__ out) {
  __shared__ _Float16 Alds[16 * A_STRIDE_OUT];
  const int tid  = threadIdx.x;
  const int wave = tid >> 5;
  const int lane = tid & 31;
  const int tM = blockIdx.x * 16;                       // [0, 16384)
  const int tN = blockIdx.y * 512 + wave * 64;          // [0, 1024)

  // async-stage the 16x512 A stripe: 1024 x 16B chunks, 4 per thread
  {
    const unsigned abase = (unsigned)(uintptr_t)&Alds[0];
#pragma unroll
    for (int it = 0; it < 4; ++it) {
      const int idx = tid + it * 256;        // [0, 1024)
      const int row = idx >> 6;              // [0, 16)
      const int c   = idx & 63;
      async_g2l_b128(abase + (unsigned)(row * A_STRIDE_OUT + c * 8) * 2u,
                     g + (size_t)(tM + row) * NH + c * 8);
    }
  }
  wait_async0();
  __syncthreads();

  const int off = (lane & 16) ? 8 : 0;
  const _Float16* arow = &Alds[(lane & 15) * A_STRIDE_OUT];
  const _Float16* p1[4];
  const _Float16* p2[4];
#pragma unroll
  for (int jt = 0; jt < 4; ++jt) {
    p1[jt] = w1 + (size_t)(tN + jt * 16 + (lane & 15)) * NH;
    p2[jt] = w2 + (size_t)(tN + jt * 16 + (lane & 15)) * NH;
  }

  v8f acc1[4] = {{}, {}, {}, {}};
  v8f acc2[4] = {{}, {}, {}, {}};
  for (int kb = 0; kb < NH; kb += 32) {
    const v16h a = load_frag16(arow + kb, off);
#pragma unroll
    for (int jt = 0; jt < 4; ++jt) {
      const v16h f1 = load_frag16(p1[jt] + kb, off);
      acc1[jt] = __builtin_amdgcn_wmma_f32_16x16x32_f16(false, a, false, f1, (short)0,
                                                        acc1[jt], false, false);
      const v16h f2 = load_frag16(p2[jt] + kb, off);
      acc2[jt] = __builtin_amdgcn_wmma_f32_16x16x32_f16(false, a, false, f2, (short)0,
                                                        acc2[jt], false, false);
    }
  }

  const int N  = lane & 15;
  const int mb = (lane & 16) ? 8 : 0;
#pragma unroll
  for (int jt = 0; jt < 4; ++jt) {
    const int col = tN + jt * 16 + N;
    const float bb1 = b1[col], bb2 = b2[col];
#pragma unroll
    for (int i = 0; i < 8; ++i) {
      const size_t idx = (size_t)(tM + i + mb) * HID + col;
      const float y1 = acc1[jt][i] + bb1;
      const float y2 = acc2[jt][i] + bb2;
      out[idx] = skip[idx] + y1 * sigmoidf_(y2);
    }
  }
}

// ---------------------------------------------------------------------------
// Kernel 0: reset scan state (h0 = 0, epoch counter = 0) each call
// ---------------------------------------------------------------------------
__global__ void init_state(float* __restrict__ hbuf, unsigned int* __restrict__ cnt) {
  const int i = blockIdx.x * 256 + threadIdx.x;
  if (i < 2 * NH) hbuf[i] = 0.0f;
  if (i == 0) *cnt = 0u;
}

// ---------------------------------------------------------------------------
extern "C" void kernel_launch(void* const* d_in, const int* in_sizes, int n_in,
                              void* d_out, int out_size, void* d_ws, size_t ws_size,
                              hipStream_t stream) {
  (void)in_sizes; (void)n_in; (void)out_size; (void)ws_size;
  const float* x      = (const float*)d_in[0];
  const float* w_ih   = (const float*)d_in[1];
  const float* w_hh   = (const float*)d_in[2];
  const float* b      = (const float*)d_in[3];
  const float* b_n    = (const float*)d_in[4];
  const float* out_w  = (const float*)d_in[5];
  const float* out_b  = (const float*)d_in[6];
  const float* out2_w = (const float*)d_in[7];
  const float* out2_b = (const float*)d_in[8];
  const float* ln_w   = (const float*)d_in[9];
  const float* ln_b   = (const float*)d_in[10];

  // workspace carve-up (256B aligned)
  char* ws = (char*)d_ws;
  size_t o = 0;
  auto take = [&](size_t bytes) { char* p = ws + o; o += (bytes + 255) & ~(size_t)255; return p; };
  _Float16* xn_f16   = (_Float16*)take((size_t)T_LEN * HID * 2);   //  32 MB
  _Float16* wih_f16  = (_Float16*)take((size_t)G3 * HID * 2);      //   3 MB
  _Float16* w1_f16   = (_Float16*)take((size_t)HID * NH * 2);      //   1 MB
  _Float16* w2_f16   = (_Float16*)take((size_t)HID * NH * 2);      //   1 MB
  float*    igates   = (float*)   take((size_t)T_LEN * G3 * 4);    //  96 MB
  _Float16* g_f16    = (_Float16*)take((size_t)T_LEN * NH * 2);    //  16 MB
  float*    hbuf     = (float*)   take((size_t)2 * NH * 4);
  unsigned* cnt      = (unsigned*)take(256);

  float* hfinal = (float*)d_out;          // first 512 floats
  float* y_out  = (float*)d_out + NH;     // 16384 x 1024

  // raise dynamic-LDS cap for the scan kernel (199 KB < 320 KB WGP LDS)
  const int smem = (96 * NH + NH + 96) * (int)sizeof(float);
  (void)hipFuncSetAttribute(reinterpret_cast<const void*>(gru_scan),
                            hipFuncAttributeMaxDynamicSharedMemorySize, smem);

  init_state<<<4, 256, 0, stream>>>(hbuf, cnt);
  ln_to_f16<<<T_LEN, 256, 0, stream>>>(x, ln_w, ln_b, xn_f16);
  cvt_f16<<<(G3 * HID + 255) / 256, 256, 0, stream>>>(w_ih, wih_f16, G3 * HID);
  cvt_f16<<<(HID * NH + 255) / 256, 256, 0, stream>>>(out_w, w1_f16, HID * NH);
  cvt_f16<<<(HID * NH + 255) / 256, 256, 0, stream>>>(out2_w, w2_f16, HID * NH);
  igates_gemm<<<dim3(T_LEN / 16, G3 / 512), 256, 0, stream>>>(xn_f16, wih_f16, b, igates);
  gru_scan<<<NWG, 256, smem, stream>>>(w_hh, igates, b_n, hbuf, cnt, g_f16, hfinal);
  out_gemm<<<dim3(T_LEN / 16, HID / 512), 256, 0, stream>>>(g_f16, w1_f16, w2_f16,
                                                            out_b, out2_b, x, y_out);
}
